// TemporalLSTMGCN_90752658964843
// MI455X (gfx1250) — matrix operand order repaired and checked by reading, not compiled, not run on hardware
//
#include <hip/hip_runtime.h>

// ---------------------------------------------------------------------------
// Types for CDNA5 WMMA (wave32): v_wmma_f32_16x16x32_bf16
// ---------------------------------------------------------------------------
typedef __attribute__((ext_vector_type(16))) __bf16           v16bf;
typedef __attribute__((ext_vector_type(16))) unsigned short   v16us;
typedef __attribute__((ext_vector_type(8)))  float            v8f;
// TDM descriptor groups (amdgpu-toolchain clang-23: 6-arg tensor_load_to_lds)
typedef __attribute__((ext_vector_type(4)))  unsigned int     u32x4;
typedef __attribute__((ext_vector_type(8)))  int              i32x8;
typedef __attribute__((ext_vector_type(4)))  int              i32x4;

#define HD   256          // H*D concat width
#define NH   4            // heads
#define DH   64           // per-head dim
#define LSTM_L 512
#define NG   32           // graphs
#define OUTD 16

static __device__ __forceinline__ unsigned short f32_to_bf16(float f) {
    unsigned int u = __float_as_uint(f);
    u += 0x7FFFu + ((u >> 16) & 1u);          // round to nearest even
    return (unsigned short)(u >> 16);
}

// order-preserving float<->uint encode for atomicMax-based segment max
static __device__ __forceinline__ unsigned int f2ord(float f) {
    unsigned int u = __float_as_uint(f);
    return (u & 0x80000000u) ? ~u : (u | 0x80000000u);
}
static __device__ __forceinline__ float ord2f(unsigned int u) {
    return __uint_as_float((u & 0x80000000u) ? (u & 0x7FFFFFFFu) : ~u);
}
#define ORD_NEG_INF 0x007FFFFFu   // f2ord(-inf)

static __device__ __forceinline__ void atomic_add_f32(float* p, float v) {
    __hip_atomic_fetch_add(p, v, __ATOMIC_RELAXED, __HIP_MEMORY_SCOPE_AGENT);
}

// ---------------------------------------------------------------------------
// fp32 -> bf16 convert (row major, unchanged layout)
// ---------------------------------------------------------------------------
__global__ void k_cvt_bf16(const float* __restrict__ in,
                           unsigned short* __restrict__ out, int n) {
    int i = blockIdx.x * blockDim.x + threadIdx.x;
    if (i < n) out[i] = f32_to_bf16(in[i]);
}

// W[K,N] fp32 -> Wt[N,K] bf16 (transpose so B fragments are contiguous)
__global__ void k_cvt_transpose_bf16(const float* __restrict__ W,
                                     unsigned short* __restrict__ Wt,
                                     int K, int Nout) {
    int i = blockIdx.x * blockDim.x + threadIdx.x;
    if (i >= K * Nout) return;
    int k = i / Nout, n = i % Nout;
    Wt[n * K + k] = f32_to_bf16(W[i]);
}

// ---------------------------------------------------------------------------
// WMMA GEMM:  C[M,Nout] (f32) = A[M,K] (bf16, row major) * Bt[Nout,K]^T (bf16)
// 8 waves / block. Block owns row-tile blockIdx.x; wave w owns col-tile
// blockIdx.y*8 + w. A tile (16 x K bf16) staged into LDS by the Tensor Data
// Mover: one 2-D D# (16 rows x K/2 dwords, row stride K/2 dwords), issued by
// wave 0, fenced with s_wait_tensorcnt 0 + block barrier.
// Fragment packing (contiguous K order, same for A and B):
//   A 16x32: lane m / m+16 hold row m, K in 0-15 / 16-31
//   B 32x16: lanes 0-15 K=0..15, lanes 16-31 K=16..31, col N = lane&15
//   C 16x16 f32: VGPR r holds M = r + 8*(lane>>4), N = lane&15
// ---------------------------------------------------------------------------
__global__ __launch_bounds__(256) void k_wmma_gemm_bf16(
    const unsigned short* __restrict__ A,
    const unsigned short* __restrict__ Bt,
    float* __restrict__ C,
    int M, int K, int Nout) {
    __shared__ unsigned int Atile[16 * 256 / 2];   // up to K=256 bf16 per row

    const int lane  = threadIdx.x;            // 0..31
    const int wave  = threadIdx.y;            // 0..7
    const int mtile = blockIdx.x;
    const int ntile = blockIdx.y * 8 + wave;

    // ---- TDM: async tensor load of the 16 x K bf16 A tile into LDS ----
    if (wave == 0) {
        unsigned long long ga =
            (unsigned long long)(const void*)A + (unsigned long long)mtile * 16 * K * 2;
        // flat->LDS aperture truncates to addr[31:0]: low 32 bits = LDS byte addr
        unsigned lds_off = (unsigned)(size_t)(void*)&Atile[0];
        unsigned kd = (unsigned)(K / 2);      // dwords per row (data_size = 4B)
        u32x4 g0;
        g0.x = 1u;                                         // count=1, user mode
        g0.y = lds_off;                                    // lds_addr
        g0.z = (unsigned)(ga & 0xFFFFFFFFu);               // global_addr[31:0]
        g0.w = (unsigned)((ga >> 32) & 0x01FFFFFFu)        // global_addr[56:32]
             | 0x80000000u;                                // type=2 ("image")
        i32x8 g1;
        g1[0] = (int)(2u << 16);                           // data_size=2 -> 4B elems
        g1[1] = (int)((kd & 0xFFFFu) << 16);               // tensor_dim0[15:0]
        g1[2] = (int)((kd >> 16) | (((unsigned)M & 0xFFFFu) << 16)); // tdim0 hi | tdim1 lo
        g1[3] = (int)(((unsigned)M >> 16) | (kd << 16));   // tdim1 hi | tile_dim0
        g1[4] = 16;                                        // tile_dim1=16, tile_dim2=0
        g1[5] = (int)kd;                                   // tensor_dim0_stride[31:0]
        g1[6] = 0;
        g1[7] = 0;
        i32x4 gz  = {0, 0, 0, 0};                          // 2-D tile: groups 2/3 unused
        i32x8 gz8 = {0, 0, 0, 0, 0, 0, 0, 0};              // extra group (clang-23 form)
        __builtin_amdgcn_tensor_load_to_lds(g0, g1, gz, gz, gz8, 0);
        __builtin_amdgcn_s_wait_tensorcnt(0);
    }
    __syncthreads();

    const int m  = lane & 15;
    const int hi = lane >> 4;
    const int nb = ntile * 16 + (lane & 15);
    const unsigned int* Brow =
        (const unsigned int*)(Bt + (size_t)nb * K) + hi * 8;

    v8f acc = {};
    const unsigned int* Ar = Atile + m * (K / 2);

    for (int kt = 0; kt < K; kt += 32) {
        union FragU { unsigned int u[8]; v16us s; };
        FragU fa, fb;
#pragma unroll
        for (int j = 0; j < 8; ++j) {
            // A pair j -> K = hi*16 + 2j  (+kt)
            int k0 = hi * 16 + 2 * j;
            fa.u[j] = Ar[(kt + k0) >> 1];
            // B pair j -> K = hi*16 + 2j (+kt); contiguous in Bt row
            fb.u[j] = Brow[(kt >> 1) + j];
        }
        if (kt + 32 < K)
            __builtin_prefetch(Brow + ((kt + 32) >> 1), 0, 3);  // near (WGP) scope

        v16bf av = __builtin_bit_cast(v16bf, fa.s);
        v16bf bv = __builtin_bit_cast(v16bf, fb.s);
        acc = __builtin_amdgcn_wmma_f32_16x16x32_bf16(
            /*neg_a=*/false, av, /*neg_b=*/false, bv,
            /*c_mod=*/(short)0, acc, /*reuse_a=*/false, /*reuse_b=*/false);
    }

    float* Crow = C + (size_t)(mtile * 16) * Nout + ntile * 16 + (lane & 15);
#pragma unroll
    for (int r = 0; r < 8; ++r)
        Crow[(size_t)(r + hi * 8) * Nout] = acc[r];
}

// ---------------------------------------------------------------------------
// attention coefficients: als/ald[n,h] = dot(hpre[n, h*64 : h*64+64], a_{s,d}[h])
// ---------------------------------------------------------------------------
__global__ void k_attn_coeff(const float* __restrict__ hpre,
                             const float* __restrict__ a_src,
                             const float* __restrict__ a_dst,
                             float* __restrict__ als,
                             float* __restrict__ ald, int n_nodes) {
    int i = blockIdx.x * blockDim.x + threadIdx.x;
    if (i >= n_nodes * NH) return;
    int node = i >> 2, hd = i & 3;
    const float* hp = hpre + (size_t)node * HD + hd * DH;
    const float* as = a_src + hd * DH;
    const float* ad = a_dst + hd * DH;
    float s = 0.f, d = 0.f;
#pragma unroll 4
    for (int k = 0; k < DH; ++k) { float v = hp[k]; s += v * as[k]; d += v * ad[k]; }
    als[i] = s;
    ald[i] = d;
}

// ---------------------------------------------------------------------------
// edge pass 1: segment max of leaky_relu(als[src]+ald[dst]) per (dst, head)
// ---------------------------------------------------------------------------
__global__ void k_edge_max(const int* __restrict__ ei,
                           const float* __restrict__ als,
                           const float* __restrict__ ald,
                           unsigned int* __restrict__ mord,
                           int E_, int n_nodes) {
    int i = blockIdx.x * blockDim.x + threadIdx.x;
    if (i >= (E_ + n_nodes) * NH) return;
    int e = i >> 2, hd = i & 3;
    int src, dst;
    if (e < E_) { src = ei[e]; dst = ei[E_ + e]; } else { src = dst = e - E_; }
    float v = als[src * NH + hd] + ald[dst * NH + hd];
    v = v > 0.f ? v : 0.2f * v;
    atomicMax(&mord[dst * NH + hd], f2ord(v));
}

// edge pass 2: p = exp(e - max); atomic segment sum
__global__ void k_edge_exp(const int* __restrict__ ei,
                           const float* __restrict__ als,
                           const float* __restrict__ ald,
                           const unsigned int* __restrict__ mord,
                           float* __restrict__ ssum,
                           float* __restrict__ pbuf,
                           int E_, int n_nodes) {
    int i = blockIdx.x * blockDim.x + threadIdx.x;
    if (i >= (E_ + n_nodes) * NH) return;
    int e = i >> 2, hd = i & 3;
    int src, dst;
    if (e < E_) { src = ei[e]; dst = ei[E_ + e]; } else { src = dst = e - E_; }
    float v = als[src * NH + hd] + ald[dst * NH + hd];
    v = v > 0.f ? v : 0.2f * v;
    float p = expf(v - ord2f(mord[dst * NH + hd]));
    pbuf[i] = p;
    atomic_add_f32(&ssum[dst * NH + hd], p);
}

// edge pass 3: one wave32 per edge; lane covers 8 of the 256 channels
__global__ __launch_bounds__(256) void k_edge_scatter(
    const int* __restrict__ ei,
    const float* __restrict__ hpre,
    const float* __restrict__ ssum,
    const float* __restrict__ pbuf,
    float* __restrict__ agg,
    int E_, int n_nodes) {
    int wid  = (blockIdx.x * blockDim.x + threadIdx.x) >> 5;
    int lane = threadIdx.x & 31;
    if (wid >= E_ + n_nodes) return;
    int src, dst;
    if (wid < E_) { src = ei[wid]; dst = ei[E_ + wid]; } else { src = dst = wid - E_; }
    float alpha[NH];
#pragma unroll
    for (int hd = 0; hd < NH; ++hd)
        alpha[hd] = pbuf[wid * NH + hd] / (ssum[dst * NH + hd] + 1e-16f);
    const float* hs = hpre + (size_t)src * HD;
    float*       ag = agg  + (size_t)dst * HD;
#pragma unroll
    for (int i = 0; i < 8; ++i) {
        int c = lane + 32 * i;
        atomic_add_f32(&ag[c], hs[c] * alpha[c >> 6]);
    }
}

// epilogue: out = relu(agg + bias); optional f32 and bf16 outputs
__global__ void k_gat_epilogue(const float* __restrict__ agg,
                               const float* __restrict__ bias,
                               float* __restrict__ outf,
                               unsigned short* __restrict__ outb,
                               int n_nodes) {
    int i = blockIdx.x * blockDim.x + threadIdx.x;
    if (i >= n_nodes * HD) return;
    float v = agg[i] + bias[i & (HD - 1)];
    v = v > 0.f ? v : 0.f;
    if (outf) outf[i] = v;
    if (outb) outb[i] = f32_to_bf16(v);
}

// fill u32 pattern (works for f32 zero / ORD_NEG_INF init)
__global__ void k_fill_u32(unsigned int* __restrict__ p, unsigned int v, int n) {
    int i = blockIdx.x * blockDim.x + threadIdx.x;
    if (i < n) p[i] = v;
}

// mean pool over batch ids
__global__ void k_pool_sum(const float* __restrict__ h2,
                           const int* __restrict__ batch,
                           float* __restrict__ pooled,
                           float* __restrict__ cnt, int n_nodes) {
    int i = blockIdx.x * blockDim.x + threadIdx.x;
    if (i >= n_nodes * HD) return;
    int node = i >> 8, c = i & (HD - 1);
    int g = batch[node];
    atomic_add_f32(&pooled[g * HD + c], h2[i]);
    if (c == 0) atomic_add_f32(&cnt[g], 1.0f);
}

__global__ void k_pool_div(float* __restrict__ pooled,
                           const float* __restrict__ cnt) {
    int i = blockIdx.x * blockDim.x + threadIdx.x;
    if (i >= NG * HD) return;
    pooled[i] /= fmaxf(cnt[i >> 8], 1.0f);
}

// LSTM gates: gates[g][j] = b_ih[j] + b_hh[j] + dot(pooled[g], W_ih[j])
__global__ void k_lstm_gates(const float* __restrict__ pooled,
                             const float* __restrict__ Wih,
                             const float* __restrict__ bih,
                             const float* __restrict__ bhh,
                             float* __restrict__ gates) {
    int i = blockIdx.x * blockDim.x + threadIdx.x;
    if (i >= NG * 4 * LSTM_L) return;
    int g = i / (4 * LSTM_L), j = i % (4 * LSTM_L);
    const float* p = pooled + g * HD;
    const float* w = Wih + (size_t)j * HD;
    float acc = bih[j] + bhh[j];
#pragma unroll 4
    for (int k = 0; k < HD; ++k) acc += p[k] * w[k];
    gates[i] = acc;
}

__global__ void k_lstm_act(const float* __restrict__ gates,
                           float* __restrict__ hh) {
    int i = blockIdx.x * blockDim.x + threadIdx.x;
    if (i >= NG * LSTM_L) return;
    int g = i / LSTM_L, l = i % LSTM_L;
    const float* gr = gates + g * 4 * LSTM_L;
    float ig = gr[l], gg = gr[2 * LSTM_L + l], og = gr[3 * LSTM_L + l];
    float c  = (1.f / (1.f + expf(-ig))) * tanhf(gg);
    hh[i]    = (1.f / (1.f + expf(-og))) * tanhf(c);
}

// vsum[d] = sum_g relu(bl[d] + dot(hh[g], Wl[:,d]))
__global__ void k_head_v(const float* __restrict__ hh,
                         const float* __restrict__ Wl,
                         const float* __restrict__ bl,
                         float* __restrict__ vsum) {
    int g = blockIdx.x;     // 0..31
    int d = threadIdx.x;    // 0..63
    const float* hr = hh + g * LSTM_L;
    float acc = bl[d];
    for (int l = 0; l < LSTM_L; ++l) acc += hr[l] * Wl[l * DH + d];
    atomic_add_f32(&vsum[d], fmaxf(acc, 0.f));
}

// out[o] = sigmoid(bf[o] + dot(vsum, Wf[:,o]))
__global__ void k_head_out(const float* __restrict__ vsum,
                           const float* __restrict__ Wf,
                           const float* __restrict__ bfb,
                           float* __restrict__ out) {
    int o = threadIdx.x;
    if (o >= OUTD) return;
    float acc = bfb[o];
    for (int d = 0; d < DH; ++d) acc += vsum[d] * Wf[d * OUTD + o];
    out[o] = 1.f / (1.f + expf(-acc));
}

// ---------------------------------------------------------------------------
// host launch
// ---------------------------------------------------------------------------
static inline size_t align256(size_t x) { return (x + 255) & ~(size_t)255; }

extern "C" void kernel_launch(void* const* d_in, const int* in_sizes, int n_in,
                              void* d_out, int out_size, void* d_ws, size_t ws_size,
                              hipStream_t stream) {
    const float* x     = (const float*)d_in[0];
    const int*   ei    = (const int*)  d_in[1];
    const int*   batch = (const int*)  d_in[2];
    const float* W1    = (const float*)d_in[3];
    const float* as1   = (const float*)d_in[4];
    const float* ad1   = (const float*)d_in[5];
    const float* b1    = (const float*)d_in[6];
    const float* W2    = (const float*)d_in[7];
    const float* as2   = (const float*)d_in[8];
    const float* ad2   = (const float*)d_in[9];
    const float* b2    = (const float*)d_in[10];
    const float* Wih   = (const float*)d_in[11];
    // d_in[12] = W_hh: unused (h0 == 0)
    const float* bih   = (const float*)d_in[13];
    const float* bhh   = (const float*)d_in[14];
    const float* Wl    = (const float*)d_in[15];
    const float* bl    = (const float*)d_in[16];
    const float* Wf    = (const float*)d_in[17];
    const float* bfb   = (const float*)d_in[18];
    float* out = (float*)d_out;

    const int FIN = 128;
    const int Nn  = in_sizes[0] / FIN;   // 10000
    const int Ee  = in_sizes[1] / 2;     // 320000
    const int Et  = Ee + Nn;             // edges + self loops

    // ---- workspace carve ----
    char* ws = (char*)d_ws;
    size_t off = 0;
    auto carve = [&](size_t bytes) { char* p = ws + off; off = align256(off + bytes); return p; };
    unsigned short* xb   = (unsigned short*)carve((size_t)Nn * FIN * 2);
    unsigned short* W1t  = (unsigned short*)carve((size_t)HD * FIN * 2);
    unsigned short* W2t  = (unsigned short*)carve((size_t)HD * HD * 2);
    unsigned short* h1b  = (unsigned short*)carve((size_t)Nn * HD * 2);
    float*          hpre = (float*)carve((size_t)Nn * HD * 4);
    float*          agg  = (float*)carve((size_t)Nn * HD * 4);
    float*          als  = (float*)carve((size_t)Nn * NH * 4);
    float*          ald  = (float*)carve((size_t)Nn * NH * 4);
    unsigned int*   mord = (unsigned int*)carve((size_t)Nn * NH * 4);
    float*          ssum = (float*)carve((size_t)Nn * NH * 4);
    float*          pbuf = (float*)carve((size_t)Et * NH * 4);
    float*          pooled = (float*)carve((size_t)NG * HD * 4);
    float*          cnt    = (float*)carve((size_t)NG * 4);
    float*          gates  = (float*)carve((size_t)NG * 4 * LSTM_L * 4);
    float*          hh     = (float*)carve((size_t)NG * LSTM_L * 4);
    float*          vsum   = (float*)carve((size_t)DH * 4);
    (void)ws_size; (void)n_in; (void)out_size;

    const int T = 256;
    auto blocks = [&](long long n) { return (unsigned)((n + T - 1) / T); };

    // ---- converts ----
    k_cvt_bf16<<<blocks((long long)Nn * FIN), T, 0, stream>>>(x, xb, Nn * FIN);
    k_cvt_transpose_bf16<<<blocks(FIN * HD), T, 0, stream>>>(W1, W1t, FIN, HD);
    k_cvt_transpose_bf16<<<blocks(HD * HD), T, 0, stream>>>(W2, W2t, HD, HD);

    dim3 gemm_blk(32, 8, 1);

    // =================== GAT layer 1 ===================
    { dim3 g(Nn / 16, HD / 128, 1);
      k_wmma_gemm_bf16<<<g, gemm_blk, 0, stream>>>(xb, W1t, hpre, Nn, FIN, HD); }
    k_attn_coeff<<<blocks(Nn * NH), T, 0, stream>>>(hpre, as1, ad1, als, ald, Nn);
    k_fill_u32<<<blocks(Nn * NH), T, 0, stream>>>(mord, ORD_NEG_INF, Nn * NH);
    k_fill_u32<<<blocks(Nn * NH), T, 0, stream>>>((unsigned int*)ssum, 0u, Nn * NH);
    k_fill_u32<<<blocks((long long)Nn * HD), T, 0, stream>>>((unsigned int*)agg, 0u, Nn * HD);
    k_edge_max<<<blocks((long long)Et * NH), T, 0, stream>>>(ei, als, ald, mord, Ee, Nn);
    k_edge_exp<<<blocks((long long)Et * NH), T, 0, stream>>>(ei, als, ald, mord, ssum, pbuf, Ee, Nn);
    k_edge_scatter<<<blocks((long long)Et * 32), T, 0, stream>>>(ei, hpre, ssum, pbuf, agg, Ee, Nn);
    k_gat_epilogue<<<blocks((long long)Nn * HD), T, 0, stream>>>(agg, b1, (float*)nullptr, h1b, Nn);

    // =================== GAT layer 2 ===================
    { dim3 g(Nn / 16, HD / 128, 1);
      k_wmma_gemm_bf16<<<g, gemm_blk, 0, stream>>>(h1b, W2t, hpre, Nn, HD, HD); }
    k_attn_coeff<<<blocks(Nn * NH), T, 0, stream>>>(hpre, as2, ad2, als, ald, Nn);
    k_fill_u32<<<blocks(Nn * NH), T, 0, stream>>>(mord, ORD_NEG_INF, Nn * NH);
    k_fill_u32<<<blocks(Nn * NH), T, 0, stream>>>((unsigned int*)ssum, 0u, Nn * NH);
    k_fill_u32<<<blocks((long long)Nn * HD), T, 0, stream>>>((unsigned int*)agg, 0u, Nn * HD);
    k_edge_max<<<blocks((long long)Et * NH), T, 0, stream>>>(ei, als, ald, mord, Ee, Nn);
    k_edge_exp<<<blocks((long long)Et * NH), T, 0, stream>>>(ei, als, ald, mord, ssum, pbuf, Ee, Nn);
    k_edge_scatter<<<blocks((long long)Et * 32), T, 0, stream>>>(ei, hpre, ssum, pbuf, agg, Ee, Nn);
    // h2 (f32, relu+bias) written back into hpre (scatter already consumed it)
    k_gat_epilogue<<<blocks((long long)Nn * HD), T, 0, stream>>>(agg, b2, hpre, (unsigned short*)nullptr, Nn);

    // =================== pool + LSTM + head ===================
    k_fill_u32<<<blocks(NG * HD), T, 0, stream>>>((unsigned int*)pooled, 0u, NG * HD);
    k_fill_u32<<<1, T, 0, stream>>>((unsigned int*)cnt, 0u, NG);
    k_pool_sum<<<blocks((long long)Nn * HD), T, 0, stream>>>(hpre, batch, pooled, cnt, Nn);
    k_pool_div<<<blocks(NG * HD), T, 0, stream>>>(pooled, cnt);
    k_lstm_gates<<<blocks(NG * 4 * LSTM_L), T, 0, stream>>>(pooled, Wih, bih, bhh, gates);
    k_lstm_act<<<blocks(NG * LSTM_L), T, 0, stream>>>(gates, hh);
    k_fill_u32<<<1, DH, 0, stream>>>((unsigned int*)vsum, 0u, DH);
    k_head_v<<<NG, DH, 0, stream>>>(hh, Wl, bl, vsum);
    k_head_out<<<1, 32, 0, stream>>>(vsum, Wf, bfb, out);
}
